// MeshMLPL2Net_3934190043219
// MI455X (gfx1250) — matrix operand order repaired and checked by reading, not compiled, not run on hardware
//
#include <hip/hip_runtime.h>
#include <hip/hip_bf16.h>
#include <stdint.h>

// ---------------------------------------------------------------------------
// MeshMLPL2Net forward on gfx1250 (MI455X, wave32, WMMA).
// Dense node-feature GEMMs use v_wmma_f32_16x16x32_bf16 (bf16 in, f32 acc):
// layer-3 GEMMs have AI ~64 FLOP/B, so the matrix pipes are required to stay
// HBM-bound at 23.3 TB/s. Each wave accumulates 4 col-tiles (A-frag reuse,
// 4x less A re-read). Edge scatter/gather stays f32 + global atomic adds
// (pure bandwidth). Exact per-graph top-k via 4-pass radix select.
// ---------------------------------------------------------------------------

typedef __attribute__((ext_vector_type(16))) __bf16        v16bf;
typedef __attribute__((ext_vector_type(8)))  float         v8f;
typedef __attribute__((ext_vector_type(4)))  unsigned int  u32x4;
typedef __attribute__((ext_vector_type(2)))  unsigned int  u32x2;

union BFrag { v16bf v; u32x4 q[2]; };

__device__ __forceinline__ unsigned short f2bf(float f) {
  unsigned int u = __float_as_uint(f);
  u += 0x7fffu + ((u >> 16) & 1u);          // round-to-nearest-even
  return (unsigned short)(u >> 16);
}

// float -> monotone uint (descending-order radix select on "larger is larger")
__device__ __forceinline__ unsigned int fflip(float f) {
  unsigned int u = __float_as_uint(f);
  return (u & 0x80000000u) ? ~u : (u | 0x80000000u);
}

// ---------------------------------------------------------------------------
// GEMM: C[n x N] = act(A[n x K] @ W[K x N] + bias). W passed as bf16
// transposed Wt[N][K]. Block = 8 waves; each wave owns 16 rows and NT
// 16-col tiles (acc[NT], one A fragment reused across NT WMMAs).
// K consumed 32 at a time through LDS (pitch 40 ushorts -> conflict-free
// b128 reads). blockIdx.x = col panel (small) so blocks sharing an A row
// panel are launch-adjacent and hit L2.
// ---------------------------------------------------------------------------
template <int NT>
__global__ void gemm_bf16_wmma(const float* __restrict__ A,
                               const unsigned short* __restrict__ Wt,
                               const float* __restrict__ bias,
                               float* __restrict__ C,
                               int n, int K, int N, int relu)
{
  __shared__ __attribute__((aligned(16))) unsigned short As[128 * 40];
  __shared__ __attribute__((aligned(16))) unsigned short Bs[16 * NT * 40];

  const int tid  = threadIdx.x;
  const int lane = tid & 31;
  const int wave = tid >> 5;
  const int row0 = blockIdx.y * 128;
  const int col0 = blockIdx.x * (16 * NT);
  const bool kfull = (K & 31) == 0;

  v8f acc[NT];
#pragma unroll
  for (int j = 0; j < NT; ++j) acc[j] = (v8f){};

  const int ksteps = (K + 31) >> 5;
  for (int ks = 0; ks < ksteps; ++ks) {
    const int k0 = ks << 5;
    if (kfull) {
      // fast path: vectorized staging, row-clamped (no zero pad needed:
      // WMMA output rows depend only on their own A row; extra rows unstored)
#pragma unroll
      for (int j = 0; j < 4; ++j) {           // 1024 float4-chunks, 4/thread
        int chunk = tid + j * 256;
        int r = chunk >> 3, cg = chunk & 7;
        int gr = row0 + r; if (gr >= n) gr = n - 1;
        float4 v = *(const float4*)(A + (size_t)gr * K + k0 + cg * 4);
        u32x2 pk;
        pk.x = ((unsigned int)f2bf(v.y) << 16) | f2bf(v.x);
        pk.y = ((unsigned int)f2bf(v.w) << 16) | f2bf(v.z);
        *(u32x2*)&As[r * 40 + cg * 4] = pk;
      }
      for (int cb = tid; cb < NT * 64; cb += 256) {   // 8-elem chunks of Wt
        int r = cb >> 2, cg = cb & 3;
        *(u32x4*)&Bs[r * 40 + cg * 8] =
            *(const u32x4*)(Wt + (size_t)(col0 + r) * K + k0 + cg * 8);
      }
    } else {
      // generic path (layer-0 mlp, K=5): scalar with zero padding
      for (int i = tid; i < 128 * 32; i += 256) {
        int r = i >> 5, c = i & 31;
        int gr = row0 + r, gc = k0 + c;
        float v = (gr < n && gc < K) ? A[(size_t)gr * K + gc] : 0.f;
        As[r * 40 + c] = f2bf(v);
      }
      for (int i = tid; i < 16 * NT * 32; i += 256) {
        int r = i >> 5, c = i & 31;
        int gc = k0 + c;
        Bs[r * 40 + c] = (gc < K) ? Wt[(size_t)(col0 + r) * K + gc]
                                  : (unsigned short)0;
      }
    }
    __syncthreads();

    const int m    = lane & 15;
    const int half = lane >> 4;
    BFrag a;
    // A frag: K chunk [h*16, h*16+16)
    const unsigned short* ap = &As[(wave * 16 + m) * 40];
    a.q[0] = *(const u32x4*)(ap + half * 16);
    a.q[1] = *(const u32x4*)(ap + half * 16 + 8);
#pragma unroll
    for (int j = 0; j < NT; ++j) {
      BFrag b;   // B frag: column m of tile j; K chunk [h*16, h*16+16)
      const unsigned short* bp = &Bs[(j * 16 + m) * 40];
      b.q[0] = *(const u32x4*)(bp + half * 16);
      b.q[1] = *(const u32x4*)(bp + half * 16 + 8);
      acc[j] = __builtin_amdgcn_wmma_f32_16x16x32_bf16(
          false, a.v, false, b.v, (short)0, acc[j], false, false);
    }
    __syncthreads();
  }

  // store: lane -> col = lane&15, rows = (lane>>4)*8 + r
  const int m0 = (lane >> 4) * 8;
#pragma unroll
  for (int j = 0; j < NT; ++j) {
    const int cc = col0 + j * 16 + (lane & 15);
    const float bv = bias ? bias[cc] : 0.f;
#pragma unroll
    for (int r = 0; r < 8; ++r) {
      int row = row0 + wave * 16 + m0 + r;
      if (row < n) {
        float v = acc[j][r] + bv;
        if (relu) v = fmaxf(v, 0.f);
        C[(size_t)row * N + cc] = v;
      }
    }
  }
}

// ---------------------------------------------------------------------------
// small utility kernels
// ---------------------------------------------------------------------------
__global__ void k_wt_bf16(const float* __restrict__ W,
                          unsigned short* __restrict__ Wt, int K, int N) {
  int i = blockIdx.x * 256 + threadIdx.x;
  if (i < K * N) {
    int r = i / N, c = i % N;
    Wt[(size_t)c * K + r] = f2bf(W[i]);
  }
}

__global__ void k_init_edges(const int* __restrict__ ei, int* __restrict__ s,
                             int* __restrict__ d, float* __restrict__ w, int E) {
  int e = blockIdx.x * 256 + threadIdx.x;
  if (e < E) { s[e] = ei[e]; d[e] = ei[E + e]; w[e] = 1.f; }
}

__global__ void k_deg(const int* __restrict__ dst, const float* __restrict__ w,
                      float* __restrict__ deg, int E) {
  int e = blockIdx.x * 256 + threadIdx.x;
  if (e < E && w[e] > 0.f) atomicAdd(&deg[dst[e]], 1.f);
}

__global__ void k_dinv(const float* __restrict__ deg, float* __restrict__ dinv, int n) {
  int i = blockIdx.x * 256 + threadIdx.x;
  if (i < n) dinv[i] = rsqrtf(deg[i] + 1.f);   // +1 self-loop
}

// out[i,c] = dinv_i^2 * h[i,c] + b[c]
__global__ void k_selfloop(const float* __restrict__ dinv,
                           const float* __restrict__ h,
                           const float* __restrict__ b,
                           float* __restrict__ out, int n, int Cc, int lc) {
  int i = blockIdx.x * 256 + threadIdx.x;
  if (i < n * Cc) {
    int node = i >> lc, c = i & (Cc - 1);
    float di = dinv[node];
    out[i] = di * di * h[i] + b[c];
  }
}

// atomic scatter of coef * h[src] into out[dst]; 4 channels per thread
__global__ void k_edge_agg(const int* __restrict__ s, const int* __restrict__ d,
                           const float* __restrict__ w, const float* __restrict__ dinv,
                           const float* __restrict__ h, float* __restrict__ out,
                           int E, int Cc, int lcpt) {
  int gid = blockIdx.x * 256 + threadIdx.x;
  int cpt = Cc >> 2;
  int e = gid >> lcpt;
  if (e < E) {
    float wv = w[e];
    if (wv > 0.f) {
      int c0 = (gid & (cpt - 1)) << 2;
      int si = s[e], di = d[e];
      float coef = dinv[si] * dinv[di];
      const float* hp = h + (size_t)si * Cc + c0;
      float* op = out + (size_t)di * Cc + c0;
      float4 hv = *(const float4*)hp;
      atomicAdd(op + 0, coef * hv.x);
      atomicAdd(op + 1, coef * hv.y);
      atomicAdd(op + 2, coef * hv.z);
      atomicAdd(op + 3, coef * hv.w);
    }
  }
}

__global__ void k_bn_stats(const float* __restrict__ x, float* __restrict__ s,
                           float* __restrict__ s2, int n, int Cc, int relu) {
  int c = threadIdx.x & (Cc - 1);
  int rpb = 256 / Cc;
  int r = blockIdx.x * rpb + (threadIdx.x / Cc);
  int stride = gridDim.x * rpb;
  float a = 0.f, b = 0.f;
  for (; r < n; r += stride) {
    float v = x[(size_t)r * Cc + c];
    if (relu) v = fmaxf(v, 0.f);
    a += v; b += v * v;
  }
  atomicAdd(&s[c], a);
  atomicAdd(&s2[c], b);
}

__global__ void k_bn_apply(const float* __restrict__ x, float* __restrict__ y,
                           const float* __restrict__ s, const float* __restrict__ s2,
                           const float* __restrict__ g, const float* __restrict__ beta,
                           int n, int Cc, int relu) {
  int i = blockIdx.x * 256 + threadIdx.x;
  if (i < n * Cc) {
    int c = i & (Cc - 1);
    float inv_n = 1.f / (float)n;
    float mu = s[c] * inv_n;
    float var = s2[c] * inv_n - mu * mu;
    float v = x[i];
    if (relu) v = fmaxf(v, 0.f);
    y[i] = g[c] * (v - mu) * rsqrtf(var + 1e-5f) + beta[c];
  }
}

// pool score: hs = x @ pw (co -> 1). one wave per node, coalesced lanes +
// shuffle reduction.
__global__ void k_pool_h(const float* __restrict__ x, const float* __restrict__ pw,
                         float* __restrict__ hs, int n, int Cc) {
  int node = blockIdx.x * 8 + (threadIdx.x >> 5);
  int lane = threadIdx.x & 31;
  if (node < n) {
    const float* xp = x + (size_t)node * Cc;
    float a = 0.f;
    for (int c = lane; c < Cc; c += 32) a += xp[c] * pw[c];
#pragma unroll
    for (int o = 16; o > 0; o >>= 1) a += __shfl_xor(a, o, 32);
    if (lane == 0) hs[node] = a;
  }
}

__global__ void k_score_init(const float* __restrict__ dinv, const float* __restrict__ hs,
                             const float* __restrict__ pb, float* __restrict__ sc, int n) {
  int i = blockIdx.x * 256 + threadIdx.x;
  if (i < n) { float di = dinv[i]; sc[i] = di * di * hs[i] + pb[0]; }
}

__global__ void k_score_edge(const int* __restrict__ s, const int* __restrict__ d,
                             const float* __restrict__ w, const float* __restrict__ dinv,
                             const float* __restrict__ hs, float* __restrict__ sc, int E) {
  int e = blockIdx.x * 256 + threadIdx.x;
  if (e < E && w[e] > 0.f)
    atomicAdd(&sc[d[e]], dinv[s[e]] * dinv[d[e]] * hs[s[e]]);
}

// exact top-k per graph: 4-pass radix select on flipped float keys.
// one block (256 threads) per graph; writes mapping[old] = new index or -1.
__global__ void k_topk(const float* __restrict__ score, int* __restrict__ mapping,
                       int n_per, int k) {
  int g = blockIdx.x;
  const float* sc = score + (size_t)g * n_per;
  int* mp = mapping + (size_t)g * n_per;
  __shared__ unsigned int hist[256];
  __shared__ unsigned int sh_prefix;
  __shared__ int sh_rem;
  __shared__ unsigned int cnt_gt, cnt_eq;

  unsigned int prefix = 0;
  int rem = k;
  for (int pass = 0; pass < 4; ++pass) {
    int shift = 24 - 8 * pass;
    hist[threadIdx.x] = 0;
    __syncthreads();
    unsigned int mask = (pass == 0) ? 0u : (0xFFFFFFFFu << (shift + 8));
    for (int i = threadIdx.x; i < n_per; i += 256) {
      unsigned int key = fflip(sc[i]);
      if ((key & mask) == (prefix & mask))
        atomicAdd(&hist[(key >> shift) & 255], 1u);
    }
    __syncthreads();
    if (threadIdx.x == 0) {
      int cum = 0, b = 255;
      for (; b >= 0; --b) { cum += (int)hist[b]; if (cum >= rem) break; }
      if (b < 0) b = 0;
      sh_prefix = prefix | ((unsigned int)b << shift);
      sh_rem = rem - (cum - (int)hist[b]);
    }
    __syncthreads();
    prefix = sh_prefix;
    rem = sh_rem;
    __syncthreads();
  }
  const unsigned int T = prefix;      // exact key of k-th largest
  const int kgt = k - rem;            // count strictly greater than T
  if (threadIdx.x == 0) { cnt_gt = 0u; cnt_eq = 0u; }
  __syncthreads();
  for (int i = threadIdx.x; i < n_per; i += 256) {
    unsigned int key = fflip(sc[i]);
    if (key > T) {
      int p = (int)atomicAdd(&cnt_gt, 1u);
      mp[i] = g * k + p;
    } else if (key == T) {
      int p = (int)atomicAdd(&cnt_eq, 1u);
      if (p < rem) mp[i] = g * k + kgt + p;
    }
  }
}

__global__ void k_gather(const float* __restrict__ x, const float* __restrict__ sc,
                         const int* __restrict__ mapping, float* __restrict__ y,
                         int n_old, int Cc, int lc) {
  int i = blockIdx.x * 256 + threadIdx.x;
  if (i < n_old * Cc) {
    int node = i >> lc;
    int ni = mapping[node];
    if (ni >= 0) y[(size_t)ni * Cc + (i & (Cc - 1))] = x[i] * tanhf(sc[node]);
  }
}

__global__ void k_relabel(int* __restrict__ s, int* __restrict__ d,
                          float* __restrict__ w, const int* __restrict__ mapping, int E) {
  int e = blockIdx.x * 256 + threadIdx.x;
  if (e < E) {
    int ns = mapping[s[e]], nd = mapping[d[e]];
    bool val = (ns >= 0) && (nd >= 0) && (w[e] > 0.f);
    s[e] = val ? ns : 0;
    d[e] = val ? nd : 0;
    w[e] = val ? 1.f : 0.f;
  }
}

__global__ void k_mean_pool(const float* __restrict__ x, float* __restrict__ emb,
                            int n_per, int Cc) {
  int g = blockIdx.x, c = threadIdx.x;      // blockDim == Cc
  float a = 0.f;
  const float* xp = x + (size_t)g * n_per * Cc + c;
  for (int r = 0; r < n_per; ++r) a += xp[(size_t)r * Cc];
  emb[g * Cc + c] = a / (float)n_per;
}

__global__ void k_fc_head(const float* __restrict__ emb,
                          const float* __restrict__ W1, const float* __restrict__ b1,
                          const float* __restrict__ W2, const float* __restrict__ b2,
                          float* __restrict__ logits, float* __restrict__ emb_out) {
  __shared__ float h1[128];
  int g = blockIdx.x, t = threadIdx.x;      // blockDim == 128
  const float* e = emb + g * 256;
  float a = b1[t];
  for (int c = 0; c < 256; ++c) a += e[c] * W1[c * 128 + t];
  h1[t] = a;
  emb_out[g * 256 + t] = e[t];
  emb_out[g * 256 + 128 + t] = e[128 + t];
  __syncthreads();
  if (t < 30) {
    float a2 = b2[t];
    for (int c = 0; c < 128; ++c) a2 += h1[c] * W2[c * 30 + t];
    logits[g * 30 + t] = a2;
  }
}

// ---------------------------------------------------------------------------
// host launcher
// ---------------------------------------------------------------------------
static inline int ilog2i(int v) { int l = 0; while ((1 << l) < v) ++l; return l; }

static void launch_gemm(const float* A, const unsigned short* Wt, const float* bias,
                        float* C, int n, int K, int N, int relu, hipStream_t stream) {
  if (N >= 64) {
    dim3 gg(N / 64, (n + 127) / 128);
    gemm_bf16_wmma<4><<<gg, 256, 0, stream>>>(A, Wt, bias, C, n, K, N, relu);
  } else {
    dim3 gg(N / 32, (n + 127) / 128);
    gemm_bf16_wmma<2><<<gg, 256, 0, stream>>>(A, Wt, bias, C, n, K, N, relu);
  }
}

extern "C" void kernel_launch(void* const* d_in, const int* in_sizes, int n_in,
                              void* d_out, int out_size, void* d_ws, size_t ws_size,
                              hipStream_t stream) {
  (void)n_in; (void)out_size; (void)ws_size;
  // input order: x, then 4x10 layer params, fc1_W, fc1_b, fc2_W, fc2_b,
  // edge_index, batch  (top-level + params dict insertion order)
  const float* x_in = (const float*)d_in[0];
  const int* edge_index = (const int*)d_in[45];
  const int E = in_sizes[45] / 2;

  const int B = 32;
  const int np[5] = {10000, 8000, 6400, 5120, 4096};
  const int ci[4] = {5, 32, 64, 128};
  const int co[4] = {32, 64, 128, 256};

  // workspace carve
  char* p = (char*)d_ws;
  auto carve = [&](size_t bytes) -> char* {
    char* r = p; p += (bytes + 255) & ~(size_t)255; return r;
  };
  const size_t MAXE = (size_t)B * 5120 * 256;   // 41,943,040 elems
  float* bufA = (float*)carve(MAXE * 4);
  float* bufB = (float*)carve(MAXE * 4);
  float* bufH = (float*)carve(MAXE * 4);
  int*   esrc = (int*)carve((size_t)E * 4);
  int*   edst = (int*)carve((size_t)E * 4);
  float* ew   = (float*)carve((size_t)E * 4);
  float* deg  = (float*)carve(320000 * 4);
  float* dinv = (float*)carve(320000 * 4);
  float* hs   = (float*)carve(320000 * 4);
  float* scor = (float*)carve(320000 * 4);
  int*   mapv = (int*)carve(320000 * 4);
  float* emb  = (float*)carve(32 * 256 * 4);
  float* bns  = (float*)carve(256 * 4);
  float* bns2 = (float*)carve(256 * 4);
  unsigned short* wtM[4];
  unsigned short* wtC[4];
  for (int i = 0; i < 4; ++i) {
    wtM[i] = (unsigned short*)carve((size_t)ci[i] * co[i] * 2);
    wtC[i] = (unsigned short*)carve((size_t)co[i] * co[i] * 2);
  }

  const int eb = (E + 255) / 256;
  k_init_edges<<<eb, 256, 0, stream>>>(edge_index, esrc, edst, ew, E);

  const float* cur = x_in;
  for (int i = 0; i < 4; ++i) {
    const int n = B * np[i], Cc = co[i], K = ci[i];
    const int lc = ilog2i(Cc);
    const int base = 1 + i * 10;
    const float* mlp_W  = (const float*)d_in[base + 0];
    const float* mlp_b  = (const float*)d_in[base + 1];
    const float* mlp_g  = (const float*)d_in[base + 2];
    const float* mlp_bt = (const float*)d_in[base + 3];
    const float* cnv_W  = (const float*)d_in[base + 4];
    const float* cnv_b  = (const float*)d_in[base + 5];
    const float* nrm_g  = (const float*)d_in[base + 6];
    const float* nrm_bt = (const float*)d_in[base + 7];
    const float* pol_W  = (const float*)d_in[base + 8];
    const float* pol_b  = (const float*)d_in[base + 9];

    float* M = (i == 0) ? bufA : ((cur == bufA) ? bufB : bufA);
    float* G = (i == 0) ? bufB : (float*)cur;

    // weights -> transposed bf16
    k_wt_bf16<<<(K * Cc + 255) / 256, 256, 0, stream>>>(mlp_W, wtM[i], K, Cc);
    k_wt_bf16<<<(Cc * Cc + 255) / 256, 256, 0, stream>>>(cnv_W, wtC[i], Cc, Cc);

    const long long nC = (long long)n * Cc;
    const unsigned int eg = (unsigned int)((nC + 255) / 256);

    // 1) MLP: relu(x@W+b) fused into gemm, then BN (stats + apply) in place
    launch_gemm(cur, wtM[i], mlp_b, M, n, K, Cc, 1, stream);
    hipMemsetAsync(bns, 0, Cc * 4, stream);
    hipMemsetAsync(bns2, 0, Cc * 4, stream);
    k_bn_stats<<<2048, 256, 0, stream>>>(M, bns, bns2, n, Cc, 0);
    k_bn_apply<<<eg, 256, 0, stream>>>(M, M, bns, bns2, mlp_g, mlp_bt, n, Cc, 0);

    // 2) GCNConv: degrees, h = x@W, self-loop + edge aggregation, relu+BN
    hipMemsetAsync(deg, 0, (size_t)n * 4, stream);
    k_deg<<<eb, 256, 0, stream>>>(edst, ew, deg, E);
    k_dinv<<<(n + 255) / 256, 256, 0, stream>>>(deg, dinv, n);
    launch_gemm(M, wtC[i], nullptr, bufH, n, Cc, Cc, 0, stream);
    k_selfloop<<<eg, 256, 0, stream>>>(dinv, bufH, cnv_b, G, n, Cc, lc);
    const int cpt = Cc >> 2, lcpt = ilog2i(cpt);
    const long long etot = (long long)E * cpt;
    k_edge_agg<<<(unsigned int)((etot + 255) / 256), 256, 0, stream>>>(
        esrc, edst, ew, dinv, bufH, G, E, Cc, lcpt);
    hipMemsetAsync(bns, 0, Cc * 4, stream);
    hipMemsetAsync(bns2, 0, Cc * 4, stream);
    k_bn_stats<<<2048, 256, 0, stream>>>(G, bns, bns2, n, Cc, 1);
    k_bn_apply<<<eg, 256, 0, stream>>>(G, G, bns, bns2, nrm_g, nrm_bt, n, Cc, 1);

    // 3) SAGPool: score = GCN(co->1), exact top-k, tanh-gated gather, relabel
    k_pool_h<<<(n + 7) / 8, 256, 0, stream>>>(G, pol_W, hs, n, Cc);
    k_score_init<<<(n + 255) / 256, 256, 0, stream>>>(dinv, hs, pol_b, scor, n);
    k_score_edge<<<eb, 256, 0, stream>>>(esrc, edst, ew, dinv, hs, scor, E);
    hipMemsetAsync(mapv, 0xFF, (size_t)n * 4, stream);
    k_topk<<<B, 256, 0, stream>>>(scor, mapv, np[i], np[i + 1]);
    k_gather<<<eg, 256, 0, stream>>>(G, scor, mapv, M, n, Cc, lc);
    k_relabel<<<eb, 256, 0, stream>>>(esrc, edst, ew, mapv, E);
    cur = M;
  }

  // head: mean pool + two linear layers; d_out = logits(32x30) || emb(32x256)
  k_mean_pool<<<B, 256, 0, stream>>>(cur, emb, np[4], 256);
  k_fc_head<<<B, 128, 0, stream>>>(emb,
                                   (const float*)d_in[41], (const float*)d_in[42],
                                   (const float*)d_in[43], (const float*)d_in[44],
                                   (float*)d_out, (float*)d_out + 32 * 30);
}
